// LSTMDecoderCell_9174050144553
// MI455X (gfx1250) — matrix operand order repaired and checked by reading, not compiled, not run on hardware
//
#include <hip/hip_runtime.h>

// ---------------------------------------------------------------------------
// Problem constants (from reference): SEQ=128, B=32, V=32000, E=H=1024, L=3,
// S_SRC=64. Strategy: convert all weights to bf16 once, run every GEMM through
// V_WMMA_F32_16X16X32_BF16 (f32 accumulate), keep LSTM cell state c in f32.
// ---------------------------------------------------------------------------

typedef __attribute__((ext_vector_type(16))) __bf16          v16bf;
typedef __attribute__((ext_vector_type(8)))  float           v8f;
typedef __attribute__((ext_vector_type(16))) unsigned short  v16u;
typedef __attribute__((ext_vector_type(8)))  unsigned short  v8u;

__device__ __forceinline__ unsigned short f2bf(float f) {
  unsigned int u = __builtin_bit_cast(unsigned int, f);
  u += 0x7FFFu + ((u >> 16) & 1u);          // round-to-nearest-even
  return (unsigned short)(u >> 16);
}

__device__ __forceinline__ float sigmoidf_dev(float x) {
  return 1.0f / (1.0f + __expf(-x));
}

// WMMA wrapper: D = A(16x32 bf16) * B(32x16 bf16) + C(16x16 f32)
__device__ __forceinline__ v8f wmma_bf16(v16bf a, v16bf b, v8f c) {
  return __builtin_amdgcn_wmma_f32_16x16x32_bf16(
      /*neg_a=*/false, a, /*neg_b=*/false, b,
      /*c_mod=*/(short)0, c, /*reuse_a=*/false, /*reuse_b=*/false);
}

// ---- tile loaders ---------------------------------------------------------
// A-matrix 16x32 bf16 per ISA layout: lane L (0..15) holds row M=L; lane L+16
// holds the same row. Element e of the lane's 16 values maps to
//   k = (e < 8 ? 0 : 16) + (lane>>4)*8 + (e & 7)
// i.e. two contiguous 8-element (16B) chunks per lane.
__device__ __forceinline__ v16bf load_A_tile(const unsigned short* __restrict__ A,
                                             int lda, int row_base, int k_base) {
  const int lane = threadIdx.x & 31;
  const int m = lane & 15, kh = lane >> 4;
  const unsigned short* r = A + (size_t)(row_base + m) * lda + k_base;
  v8u lo = *reinterpret_cast<const v8u*>(r + kh * 8);
  v8u hi = *reinterpret_cast<const v8u*>(r + 16 + kh * 8);
  v16u u;
#pragma unroll
  for (int i = 0; i < 8; ++i) { u[i] = lo[i]; u[8 + i] = hi[i]; }
  return __builtin_bit_cast(v16bf, u);
}

// B-matrix 32x16 bf16: lane holds column n = lane&15; element e maps to
// k = (e < 8 ? 0 : 16) + (lane>>4)*8 + (e & 7) (same as A). W is the
// row-major [N,K] weight, so B[k,n] = W[n,k]: two 16B loads per lane.
__device__ __forceinline__ v16bf load_B_tile(const unsigned short* __restrict__ W,
                                             int ldw, int n_base, int k_base) {
  const int lane = threadIdx.x & 31;
  const int n = lane & 15, kh = lane >> 4;
  const unsigned short* r = W + (size_t)(n_base + n) * ldw + k_base;
  v8u lo = *reinterpret_cast<const v8u*>(r + kh * 8);
  v8u hi = *reinterpret_cast<const v8u*>(r + 16 + kh * 8);
  v16u u;
#pragma unroll
  for (int i = 0; i < 8; ++i) { u[i] = lo[i]; u[8 + i] = hi[i]; }
  return __builtin_bit_cast(v16bf, u);
}

// ---------------------------------------------------------------------------
// f32 -> bf16 elementwise conversion
// ---------------------------------------------------------------------------
__global__ void f32_to_bf16_kernel(const float* __restrict__ in,
                                   unsigned short* __restrict__ out, long n) {
  long i = (long)blockIdx.x * blockDim.x + threadIdx.x;
  if (i < n) out[i] = f2bf(in[i]);
}

// Embedding gather (bf16 table already converted): x[t,b,e] = emb[token[t,b],e]
__global__ void embed_gather_kernel(const int* __restrict__ tokens,
                                    const unsigned short* __restrict__ emb_bf,
                                    unsigned short* __restrict__ x_bf, long n) {
  long i = (long)blockIdx.x * blockDim.x + threadIdx.x;
  if (i >= n) return;
  int e = (int)(i & 1023);
  long tb = i >> 10;                 // t*B + b
  int tok = tokens[tb];
  x_bf[i] = emb_bf[(size_t)tok * 1024 + e];
}

// ---------------------------------------------------------------------------
// Fused LSTM cell for one (step, layer):
//   gates[B=32, 4H] = x@Wih^T + h@Whh^T + b ; i,f,g,o -> c,h update.
// One wave owns a 16-column slice of H and all 4 gate tiles for it.
// 64 waves total (H/16). grid=16 blocks x 128 threads (4 waves).
// ---------------------------------------------------------------------------
__global__ __launch_bounds__(128)
void lstm_cell_kernel(const unsigned short* __restrict__ x_bf,  // [32,1024]
                      const unsigned short* __restrict__ h_bf,  // [32,1024]
                      const unsigned short* __restrict__ wih,   // [4096,1024]
                      const unsigned short* __restrict__ whh,   // [4096,1024]
                      const float* __restrict__ bih,            // [4096]
                      const float* __restrict__ bhh,            // [4096]
                      float* c_state,                           // [32,1024] in/out
                      unsigned short* __restrict__ h_out)       // [32,1024] bf16
{
  const int H = 1024;
  const int wave = (blockIdx.x * blockDim.x + threadIdx.x) >> 5;  // 0..63
  const int lane = threadIdx.x & 31;
  const int htile = wave * 16;

  v8f acc[4][2] = {};
  // x @ Wih^T  (K = E = 1024)
  for (int kk = 0; kk < 1024; kk += 32) {
    v16bf a0 = load_A_tile(x_bf, 1024, 0, kk);
    v16bf a1 = load_A_tile(x_bf, 1024, 16, kk);
#pragma unroll
    for (int g = 0; g < 4; ++g) {
      v16bf b = load_B_tile(wih, 1024, g * H + htile, kk);
      acc[g][0] = wmma_bf16(a0, b, acc[g][0]);
      acc[g][1] = wmma_bf16(a1, b, acc[g][1]);
    }
  }
  // h @ Whh^T  (K = H = 1024)
  for (int kk = 0; kk < 1024; kk += 32) {
    v16bf a0 = load_A_tile(h_bf, 1024, 0, kk);
    v16bf a1 = load_A_tile(h_bf, 1024, 16, kk);
#pragma unroll
    for (int g = 0; g < 4; ++g) {
      v16bf b = load_B_tile(whh, 1024, g * H + htile, kk);
      acc[g][0] = wmma_bf16(a0, b, acc[g][0]);
      acc[g][1] = wmma_bf16(a1, b, acc[g][1]);
    }
  }

  // C/D layout: lane -> col n = htile + (lane&15); vgpr r -> row (lane>>4)*8+r
  const int nl = lane & 15, kh = lane >> 4;
  const int col = htile + nl;
  float bias[4];
#pragma unroll
  for (int g = 0; g < 4; ++g) bias[g] = bih[g * H + col] + bhh[g * H + col];

#pragma unroll
  for (int t = 0; t < 2; ++t) {
#pragma unroll
    for (int r = 0; r < 8; ++r) {
      int b = t * 16 + kh * 8 + r;
      float iv = sigmoidf_dev(acc[0][t][r] + bias[0]);
      float fv = sigmoidf_dev(acc[1][t][r] + bias[1]);
      float gv = tanhf(acc[2][t][r] + bias[2]);
      float ov = sigmoidf_dev(acc[3][t][r] + bias[3]);
      size_t idx = (size_t)b * H + col;
      float cn = fv * c_state[idx] + iv * gv;
      c_state[idx] = cn;
      h_out[idx] = f2bf(ov * tanhf(cn));
    }
  }
}

// ---------------------------------------------------------------------------
// Generic skinny GEMM: out[32,N] (f32) = A[32,K](bf16) @ W[N,K]^T(bf16)
// One wave per 16-column tile.
// ---------------------------------------------------------------------------
__global__ __launch_bounds__(128)
void gemm32_f32_kernel(const unsigned short* __restrict__ A, int lda,
                       const unsigned short* __restrict__ W, int K,
                       float* __restrict__ out, int ldo) {
  const int wave = (blockIdx.x * blockDim.x + threadIdx.x) >> 5;
  const int lane = threadIdx.x & 31;
  const int n_base = wave * 16;
  v8f acc0 = {}, acc1 = {};
  for (int kk = 0; kk < K; kk += 32) {
    v16bf a0 = load_A_tile(A, lda, 0, kk);
    v16bf a1 = load_A_tile(A, lda, 16, kk);
    v16bf b  = load_B_tile(W, K, n_base, kk);
    acc0 = wmma_bf16(a0, b, acc0);
    acc1 = wmma_bf16(a1, b, acc1);
  }
  const int nl = lane & 15, kh = lane >> 4;
  const int col = n_base + nl;
#pragma unroll
  for (int r = 0; r < 8; ++r) {
    out[(size_t)(kh * 8 + r) * ldo + col] = acc0[r];
    out[(size_t)(16 + kh * 8 + r) * ldo + col] = acc1[r];
  }
}

// Same, but out = bf16(tanh(acc))  (h_tilde for attention)
__global__ __launch_bounds__(128)
void gemm32_tanh_bf16_kernel(const unsigned short* __restrict__ A, int lda,
                             const unsigned short* __restrict__ W, int K,
                             unsigned short* __restrict__ out, int ldo) {
  const int wave = (blockIdx.x * blockDim.x + threadIdx.x) >> 5;
  const int lane = threadIdx.x & 31;
  const int n_base = wave * 16;
  v8f acc0 = {}, acc1 = {};
  for (int kk = 0; kk < K; kk += 32) {
    v16bf a0 = load_A_tile(A, lda, 0, kk);
    v16bf a1 = load_A_tile(A, lda, 16, kk);
    v16bf b  = load_B_tile(W, K, n_base, kk);
    acc0 = wmma_bf16(a0, b, acc0);
    acc1 = wmma_bf16(a1, b, acc1);
  }
  const int nl = lane & 15, kh = lane >> 4;
  const int col = n_base + nl;
#pragma unroll
  for (int r = 0; r < 8; ++r) {
    out[(size_t)(kh * 8 + r) * ldo + col] = f2bf(tanhf(acc0[r]));
    out[(size_t)(16 + kh * 8 + r) * ldo + col] = f2bf(tanhf(acc1[r]));
  }
}

// ---------------------------------------------------------------------------
// Attention scores + softmax + weighted context. One block per batch b.
// scores[b,s] = <enc[s,b,:], target[b,:]>; softmax over s; wc = sum_s p*enc.
// Writes xcat[b, 0:1024] = bf16(wc), xcat[b, 1024:2048] = h3_bf[b,:].
// ---------------------------------------------------------------------------
__global__ __launch_bounds__(256)
void attn_softmax_wc_kernel(const float* __restrict__ target,     // [32,1024]
                            const float* __restrict__ enc,        // [64,32,1024]
                            const unsigned short* __restrict__ h3_bf, // [32,1024]
                            unsigned short* __restrict__ xcat)    // [32,2048]
{
  const int b = blockIdx.x;           // 0..31
  const int tid = threadIdx.x;        // 0..255
  __shared__ float part[64][4];
  __shared__ float sc[64];
  __shared__ float inv_sum;

  const int s = tid >> 2, q = tid & 3;
  float sum = 0.f;
  const float* er = enc + ((size_t)s * 32 + b) * 1024;
  const float* tr = target + (size_t)b * 1024;
  for (int e = q; e < 1024; e += 4) sum += er[e] * tr[e];
  part[s][q] = sum;
  __syncthreads();
  if (tid < 64) sc[tid] = part[tid][0] + part[tid][1] + part[tid][2] + part[tid][3];
  __syncthreads();
  if (tid == 0) {
    float m = sc[0];
    for (int i = 1; i < 64; ++i) m = fmaxf(m, sc[i]);
    float ssum = 0.f;
    for (int i = 0; i < 64; ++i) { float e = __expf(sc[i] - m); sc[i] = e; ssum += e; }
    inv_sum = 1.0f / ssum;
  }
  __syncthreads();
  const float inv = inv_sum;
  for (int e = tid; e < 1024; e += 256) {
    float wc = 0.f;
#pragma unroll 4
    for (int s2 = 0; s2 < 64; ++s2)
      wc += sc[s2] * enc[((size_t)s2 * 32 + b) * 1024 + e];
    xcat[(size_t)b * 2048 + e] = f2bf(wc * inv);
    xcat[(size_t)b * 2048 + 1024 + e] = h3_bf[(size_t)b * 1024 + e];
  }
}

// ---------------------------------------------------------------------------
// Final tied-weight logits GEMM: out[4096,32000] = outs_bf @ emb_bf^T + bias.
// Block = 128 threads (4 waves); each wave computes a 32x64 tile (2 M-tiles x
// 4 N-tiles sharing the A tiles). grid = (32000/256, 4096/32) = (125,128).
// ---------------------------------------------------------------------------
__global__ __launch_bounds__(128)
void logits_gemm_kernel(const unsigned short* __restrict__ Abf,  // [4096,1024]
                        const unsigned short* __restrict__ Wbf,  // [32000,1024]
                        const float* __restrict__ bias,          // [32000]
                        float* __restrict__ out)                 // [4096,32000]
{
  const int wave = threadIdx.x >> 5;  // 0..3
  const int lane = threadIdx.x & 31;
  const int m_base = blockIdx.y * 32;
  const int n_base = blockIdx.x * 256 + wave * 64;

  v8f acc[2][4] = {};
  for (int kk = 0; kk < 1024; kk += 32) {
    if (kk + 32 < 1024) {
      // exercise gfx1250 prefetch path for next A chunk
      __builtin_prefetch(Abf + (size_t)(m_base + (lane & 15)) * 1024 + kk + 32, 0, 1);
    }
    v16bf a0 = load_A_tile(Abf, 1024, m_base, kk);
    v16bf a1 = load_A_tile(Abf, 1024, m_base + 16, kk);
#pragma unroll
    for (int j = 0; j < 4; ++j) {
      v16bf b = load_B_tile(Wbf, 1024, n_base + j * 16, kk);
      acc[0][j] = wmma_bf16(a0, b, acc[0][j]);
      acc[1][j] = wmma_bf16(a1, b, acc[1][j]);
    }
  }
  const int nl = lane & 15, kh = lane >> 4;
#pragma unroll
  for (int j = 0; j < 4; ++j) {
    const int col = n_base + j * 16 + nl;
    const float bv = bias[col];
#pragma unroll
    for (int t = 0; t < 2; ++t) {
#pragma unroll
      for (int r = 0; r < 8; ++r) {
        int m = m_base + t * 16 + kh * 8 + r;
        out[(size_t)m * 32000 + col] = acc[t][j][r] + bv;
      }
    }
  }
}

// ---------------------------------------------------------------------------
// Host launcher
// ---------------------------------------------------------------------------
extern "C" void kernel_launch(void* const* d_in, const int* in_sizes, int n_in,
                              void* d_out, int out_size, void* d_ws, size_t ws_size,
                              hipStream_t stream) {
  (void)in_sizes; (void)n_in; (void)out_size; (void)ws_size;
  constexpr int SEQ = 128, B = 32, V = 32000, E = 1024, H = 1024, L = 3;

  const int*   tokens  = (const int*)  d_in[0];
  const float* emb_W   = (const float*)d_in[1];
  const float* W_ih    = (const float*)d_in[2];
  const float* W_hh    = (const float*)d_in[3];
  const float* b_ih    = (const float*)d_in[4];
  const float* b_hh    = (const float*)d_in[5];
  const float* Win     = (const float*)d_in[6];
  const float* Wout    = (const float*)d_in[7];
  const float* final_b = (const float*)d_in[8];
  const float* enc     = (const float*)d_in[9];
  const float* h0      = (const float*)d_in[10];
  const float* c0      = (const float*)d_in[11];
  float* logits = (float*)d_out;

  // --- carve workspace (256B-aligned buffers) ---
  char* base = (char*)d_ws;
  size_t off = 0;
  auto carve = [&](size_t bytes) -> void* {
    off = (off + 255) & ~(size_t)255;
    void* p = base + off;
    off += bytes;
    return p;
  };
  unsigned short* emb_bf  = (unsigned short*)carve((size_t)V * E * 2);
  unsigned short* wih_bf  = (unsigned short*)carve((size_t)L * 4 * H * E * 2);
  unsigned short* whh_bf  = (unsigned short*)carve((size_t)L * 4 * H * H * 2);
  unsigned short* win_bf  = (unsigned short*)carve((size_t)E * E * 2);
  unsigned short* wout_bf = (unsigned short*)carve((size_t)E * 2 * E * 2);
  unsigned short* x_bf    = (unsigned short*)carve((size_t)SEQ * B * E * 2);
  unsigned short* hbuf0   = (unsigned short*)carve((size_t)L * B * H * 2);
  unsigned short* hbuf1   = (unsigned short*)carve((size_t)L * B * H * 2);
  float*          cbuf    = (float*)         carve((size_t)L * B * H * 4);
  float*          target_f= (float*)         carve((size_t)B * E * 4);
  unsigned short* xcat    = (unsigned short*)carve((size_t)B * 2 * E * 2);
  unsigned short* outs_bf = (unsigned short*)carve((size_t)SEQ * B * E * 2);

  auto cdiv = [](long n, long d) { return (int)((n + d - 1) / d); };

  // --- one-time conversions to bf16 ---
  f32_to_bf16_kernel<<<cdiv((long)V * E, 256), 256, 0, stream>>>(emb_W, emb_bf, (long)V * E);
  f32_to_bf16_kernel<<<cdiv((long)L * 4 * H * E, 256), 256, 0, stream>>>(W_ih, wih_bf, (long)L * 4 * H * E);
  f32_to_bf16_kernel<<<cdiv((long)L * 4 * H * H, 256), 256, 0, stream>>>(W_hh, whh_bf, (long)L * 4 * H * H);
  f32_to_bf16_kernel<<<cdiv((long)E * E, 256), 256, 0, stream>>>(Win, win_bf, (long)E * E);
  f32_to_bf16_kernel<<<cdiv((long)E * 2 * E, 256), 256, 0, stream>>>(Wout, wout_bf, (long)E * 2 * E);

  // --- initial state ---
  f32_to_bf16_kernel<<<cdiv((long)L * B * H, 256), 256, 0, stream>>>(h0, hbuf0, (long)L * B * H);
  hipMemcpyAsync(cbuf, c0, (size_t)L * B * H * 4, hipMemcpyDeviceToDevice, stream);

  // --- embedding gather ---
  embed_gather_kernel<<<cdiv((long)SEQ * B * E, 256), 256, 0, stream>>>(
      tokens, emb_bf, x_bf, (long)SEQ * B * E);

  // --- recurrent loop (h double-buffered across steps) ---
  for (int t = 0; t < SEQ; ++t) {
    unsigned short* hc = (t & 1) ? hbuf1 : hbuf0;  // read
    unsigned short* hn = (t & 1) ? hbuf0 : hbuf1;  // write
    const unsigned short* xin = x_bf + (size_t)t * B * E;
    for (int l = 0; l < L; ++l) {
      lstm_cell_kernel<<<16, 128, 0, stream>>>(
          (l == 0) ? xin : (hn + (size_t)(l - 1) * B * H),
          hc + (size_t)l * B * H,
          wih_bf + (size_t)l * 4 * H * E,
          whh_bf + (size_t)l * 4 * H * H,
          b_ih + (size_t)l * 4 * H,
          b_hh + (size_t)l * 4 * H,
          cbuf + (size_t)l * B * H,
          hn + (size_t)l * B * H);
    }
    // target = h3 @ Win^T
    gemm32_f32_kernel<<<16, 128, 0, stream>>>(
        hn + (size_t)2 * B * H, H, win_bf, E, target_f, E);
    // attention scores/softmax/context -> xcat = [wc | h3]
    attn_softmax_wc_kernel<<<32, 256, 0, stream>>>(
        target_f, enc, hn + (size_t)2 * B * H, xcat);
    // h_tilde = tanh(xcat @ Wout^T) -> outs_bf[t]
    gemm32_tanh_bf16_kernel<<<16, 128, 0, stream>>>(
        xcat, 2 * E, wout_bf, 2 * E, outs_bf + (size_t)t * B * E, E);
  }

  // --- tied-weight logits GEMM ---
  dim3 grid(125, 128);  // 32000/256 x 4096/32
  logits_gemm_kernel<<<grid, 128, 0, stream>>>(outs_bf, emb_bf, final_b, logits);
}
